// LanguageModel_80333068304653
// MI455X (gfx1250) — matrix operand order repaired and checked by reading, not compiled, not run on hardware
//
#include <hip/hip_runtime.h>
#include <hip/hip_bf16.h>

// ---------------------------------------------------------------------------
// BERT-style LM forward for MI455X (gfx1250, wave32, WMMA).
// B=8 S=512 V=21128 H=768 L=12 NH=12 DH=64 DF=3072 MP=512
//
// All heavy math on the matrix pipe (v_wmma_f32_16x16x32_f16, f32 acc):
//   * QKV / O / FFN GEMMs (tiled, swizzled-B LDS, b128 fragment loads)
//   * attention QK^T and P*V
//   * relative-position scores via band-GEMM:
//       rel[q,k] = dist[q-k+511] depends only on the diagonal, so per
//       (q-tile, k-tile) the needed dist rows form an 80-row band Db.
//       Pt = Q . Db^T, Pk = K . Db^T  (both WMMA);
//       score[m,nk] = (QK[m,nk] + Pt[m,m-nk+63] + Pk[nk,m-nk+63]) * scale.
//
// B-fragments of transposed operands (K^T, Db^T, V^T) are read from
// row-major [col][kdim] LDS: each lane's 16 halves are contiguous ->
// 2x ds_load_b128, no swizzle buffers needed in attention.
//
// GEMM fuse mode is a template parameter so bias/GELU/residual epilogues are
// branch-free at ISA level.
// ---------------------------------------------------------------------------

#define Bc  8
#define Sc  512
#define Vc  21128
#define Hc  768
#define Lc  12
#define NHc 12
#define DHc 64
#define DFc 3072
#define Mrows (Bc * Sc)   // 4096

typedef __attribute__((ext_vector_type(4)))  _Float16 v4h;
typedef __attribute__((ext_vector_type(8)))  _Float16 v8h;
typedef __attribute__((ext_vector_type(16))) _Float16 v16h;
typedef __attribute__((ext_vector_type(8)))  float    v8f;

static __device__ const v8f kZero8 = {0.f, 0.f, 0.f, 0.f, 0.f, 0.f, 0.f, 0.f};

__device__ inline v8f wmma32(v16h a, v16h b, v8f c) {
  return __builtin_amdgcn_wmma_f32_16x16x32_f16(false, a, false, b,
                                                (short)0, c, false, false);
}

// ---- A fragment (16x32 f16): lane&15 = row M, lane>>4 selects K-half.
// One contiguous 16-half run per lane (k0 + 16*(lane>>4) + 0..15). ld, k0 multiples of 8.
__device__ inline v16h frag_a(const _Float16* base, int ld, int row0, int k0) {
  const int lane = threadIdx.x & 31;
  const _Float16* p = base + (size_t)(row0 + (lane & 15)) * ld + k0
                           + (((lane >> 4) & 1) << 4);
  const v8h lo = *(const v8h*)(p);
  const v8h hi = *(const v8h*)(p + 8);
  return __builtin_shufflevector(lo, hi, 0, 1, 2, 3, 4, 5, 6, 7,
                                 8, 9, 10, 11, 12, 13, 14, 15);
}

// ---- B fragment where B = T^T and T is stored row-major [col][kdim]:
// lane l -> col0+(l&15); K-run k0 + ((l>>4)<<4) + 0..15 is contiguous.
__device__ inline v16h frag_bT(const _Float16* T, int ld, int k0, int col0) {
  const int lane = threadIdx.x & 31;
  const _Float16* p = T + (size_t)(col0 + (lane & 15)) * ld + k0
                        + (((lane >> 4) & 1) << 4);
  const v8h lo = *(const v8h*)(p);
  const v8h hi = *(const v8h*)(p + 8);
  return __builtin_shufflevector(lo, hi, 0, 1, 2, 3, 4, 5, 6, 7,
                                 8, 9, 10, 11, 12, 13, 14, 15);
}

// ---- B fragment from pre-swizzled [32][16] tile (used by GEMM kernel).
__device__ inline v16h frag_b_sw(const _Float16* bf) {
  const int lane = threadIdx.x & 31;
  const _Float16* p = bf + (lane << 4);
  const v8h lo = *(const v8h*)(p);
  const v8h hi = *(const v8h*)(p + 8);
  return __builtin_shufflevector(lo, hi, 0, 1, 2, 3, 4, 5, 6, 7,
                                 8, 9, 10, 11, 12, 13, 14, 15);
}
__device__ inline void store_b_sw(_Float16* bf, int kdim, int col, _Float16 v) {
  const int j = col >> 4, n = col & 15;
  const int lane = n + ((kdim >> 4) << 4);
  bf[(((j << 5) + lane) << 4) + (kdim & 15)] = v;
}

// ---------------------------------------------------------------------------
// Generic tiled GEMM: C[M,N] = fuse(A[M,K]*B[K,N] + bias [+residual])
// FUSE: 0 = bias only, 1 = bias + exact GELU, 2 = bias + residual.
// BM=BN=64, BK=32, 128 thr / 4 waves. Template -> branch-free epilogue.
// ---------------------------------------------------------------------------
template <int FUSE>
__global__ __launch_bounds__(128) void gemm_wmma_kernel(
    const float* __restrict__ A, const float* __restrict__ Bm,
    const float* __restrict__ bias, const float* __restrict__ res,
    float* __restrict__ C, int M, int N, int K) {
  __shared__ __align__(16) _Float16 sA[64][40];
  __shared__ __align__(16) _Float16 sBf[4][32][16];
  const int tid  = threadIdx.x;
  const int wave = tid >> 5;
  const int lane = tid & 31;
  const int m0 = blockIdx.y * 64;
  const int n0 = blockIdx.x * 64;

  v8f acc[4] = {kZero8, kZero8, kZero8, kZero8};

  for (int kt = 0; kt < K; kt += 32) {
#pragma unroll
    for (int i = 0; i < 4; ++i) {
      const int e = (tid + i * 128) << 2;
      const int r = e >> 5, c = e & 31;
      const float4 f = *(const float4*)&A[(size_t)(m0 + r) * K + kt + c];
      v4h hv = {(_Float16)f.x, (_Float16)f.y, (_Float16)f.z, (_Float16)f.w};
      *(v4h*)&sA[r][c] = hv;
    }
#pragma unroll
    for (int i = 0; i < 4; ++i) {
      const int e = (tid + i * 128) << 2;
      const int r = e >> 6, c = e & 63;
      const float4 f = *(const float4*)&Bm[(size_t)(kt + r) * N + n0 + c];
      store_b_sw(&sBf[0][0][0], r, c + 0, (_Float16)f.x);
      store_b_sw(&sBf[0][0][0], r, c + 1, (_Float16)f.y);
      store_b_sw(&sBf[0][0][0], r, c + 2, (_Float16)f.z);
      store_b_sw(&sBf[0][0][0], r, c + 3, (_Float16)f.w);
    }
    __syncthreads();
    const v16h fa = frag_a(&sA[0][0], 40, wave * 16, 0);
#pragma unroll
    for (int j = 0; j < 4; ++j) {
      const v16h fb = frag_b_sw(&sBf[j][0][0]);
      acc[j] = wmma32(fa, fb, acc[j]);
    }
    __syncthreads();
  }

  const int rbase = m0 + wave * 16 + ((lane >> 4) << 3);
#pragma unroll
  for (int j = 0; j < 4; ++j) {
    const int col = n0 + j * 16 + (lane & 15);
    const float bval = bias[col];
#pragma unroll
    for (int r = 0; r < 8; ++r) {
      const size_t off = (size_t)(rbase + r) * N + col;
      float val = acc[j][r] + bval;
      if constexpr (FUSE == 1)
        val = 0.5f * val * (1.0f + erff(val * 0.70710678118654752f));
      if constexpr (FUSE == 2)
        val += res[off];
      C[off] = val;
    }
  }
}

// ---------------------------------------------------------------------------
// Fused attention per (b, head, 16-row q tile). 128 threads (4 waves).
// LDS ~54KB (CU-mode safe).
// ---------------------------------------------------------------------------
__global__ __launch_bounds__(128) void attn_kernel(
    const float* __restrict__ q, const float* __restrict__ k,
    const float* __restrict__ v, const float* __restrict__ dist,
    const int* __restrict__ x, float* __restrict__ ctx) {
  __shared__ __align__(16) _Float16 sc[16][520];   // score/prob tile (f16)
  __shared__ __align__(16) _Float16 qA[16][72];    // Q tile [m][d]
  __shared__ __align__(16) _Float16 kA[64][72];    // K [kk][d] (ph1) / V^T [d][kk] (ph3)
  __shared__ __align__(16) _Float16 Db[80][72];    // dist band [j][d]
  __shared__ __align__(16) _Float16 Pt[16][88];    // Q.Db^T
  __shared__ __align__(16) _Float16 Pk[64][88];    // K.Db^T
  __shared__ float red[128];
  __shared__ unsigned char fl[Sc];

  const int tid = threadIdx.x, wave = tid >> 5, lane = tid & 31;
  const int q0 = blockIdx.x * 16;
  const int hh = blockIdx.y;
  const int bb = blockIdx.z;
  const float* qp = q + (size_t)bb * Sc * Hc + hh * DHc;
  const float* kp = k + (size_t)bb * Sc * Hc + hh * DHc;
  const float* vp = v + (size_t)bb * Sc * Hc + hh * DHc;

  for (int i = tid; i < Sc; i += 128) fl[i] = (x[bb * Sc + i] != 0);
#pragma unroll
  for (int i = 0; i < 2; ++i) {                    // Q tile: 256 float4
    const int e = (tid + i * 128) << 2;
    const int r = e >> 6, d = e & 63;
    const float4 f = *(const float4*)&qp[(size_t)(q0 + r) * Hc + d];
    v4h hv = {(_Float16)f.x, (_Float16)f.y, (_Float16)f.z, (_Float16)f.w};
    *(v4h*)&qA[r][d] = hv;
  }
  __syncthreads();

  const int rb = (lane >> 4) << 3;                 // lane's first row (0 or 8)
  const float scale = 0.125f;                      // 1/sqrt(64)

  // ---- phase 1: score tiles ------------------------------------------------
  for (int kt = 0; kt < Sc; kt += 64) {
#pragma unroll
    for (int i = 0; i < 8; ++i) {                  // K tile: 1024 float4
      const int e = (tid + i * 128) << 2;
      const int kk = e >> 6, d = e & 63;
      const float4 f = *(const float4*)&kp[(size_t)(kt + kk) * Hc + d];
      v4h hv = {(_Float16)f.x, (_Float16)f.y, (_Float16)f.z, (_Float16)f.w};
      *(v4h*)&kA[kk][d] = hv;
    }
    const int base = q0 - kt + 448;                // dist row for band j=0
#pragma unroll
    for (int i = 0; i < 10; ++i) {                 // Db: 80x64 = 1280 float4
      const int e = (tid + i * 128) << 2;
      const int j = e >> 6, d = e & 63;
      int idx = base + j; idx = idx < 0 ? 0 : (idx > 1022 ? 1022 : idx);
      const float4 f = *(const float4*)&dist[(size_t)idx * DHc + d];
      v4h hv = {(_Float16)f.x, (_Float16)f.y, (_Float16)f.z, (_Float16)f.w};
      *(v4h*)&Db[j][d] = hv;
    }
    __syncthreads();

    // QK^T: wave w -> 16x16 tile at cols w*16
    v8f aqk = kZero8;
#pragma unroll
    for (int kc = 0; kc < 2; ++kc) {
      const v16h fa = frag_a(&qA[0][0], 72, 0, kc * 32);
      const v16h fb = frag_bT(&kA[0][0], 72, kc * 32, wave * 16);
      aqk = wmma32(fa, fb, aqk);
    }

    // Pt = Q . Db^T (16x80): 5 col-subtiles round-robin over waves
    for (int js = wave; js < 5; js += 4) {
      v8f ap = kZero8;
#pragma unroll
      for (int kc = 0; kc < 2; ++kc) {
        const v16h fa = frag_a(&qA[0][0], 72, 0, kc * 32);
        const v16h fb = frag_bT(&Db[0][0], 72, kc * 32, js * 16);
        ap = wmma32(fa, fb, ap);
      }
      const int col = js * 16 + (lane & 15);
#pragma unroll
      for (int r = 0; r < 8; ++r) Pt[rb + r][col] = (_Float16)ap[r];
    }

    // Pk = K . Db^T (64x80): 4x5 = 20 tiles round-robin over waves
    for (int t = wave; t < 20; t += 4) {
      const int rs = t / 5, js = t % 5;
      v8f ap = kZero8;
#pragma unroll
      for (int kc = 0; kc < 2; ++kc) {
        const v16h fa = frag_a(&kA[0][0], 72, rs * 16, kc * 32);
        const v16h fb = frag_bT(&Db[0][0], 72, kc * 32, js * 16);
        ap = wmma32(fa, fb, ap);
      }
      const int col = js * 16 + (lane & 15);
#pragma unroll
      for (int r = 0; r < 8; ++r) Pk[rs * 16 + rb + r][col] = (_Float16)ap[r];
    }
    __syncthreads();

    // epilogue: combine, mask, scale -> sc (f16)
    {
      const int nk = wave * 16 + (lane & 15);      // col in this 64-k tile
      const int kg = kt + nk;
#pragma unroll
      for (int r = 0; r < 8; ++r) {
        const int m  = rb + r;
        const int qg = q0 + m;
        float val;
        if (kg > qg || !fl[kg]) {
          val = -30000.0f;
        } else {
          const int j = m - nk + 63;               // band index, 0..78
          val = (aqk[r] + (float)Pt[m][j] + (float)Pk[nk][j]) * scale;
        }
        sc[m][kg] = (_Float16)val;
      }
    }
    __syncthreads();
  }

  // ---- phase 2: softmax (8 threads per row) --------------------------------
  {
    const int m = tid >> 3, part = tid & 7;
    const int i0 = part * 64;
    float mx = -1e30f;
#pragma unroll
    for (int ch = 0; ch < 8; ++ch) {
      const v8h cv = *(const v8h*)&sc[m][i0 + ch * 8];
#pragma unroll
      for (int t = 0; t < 8; ++t) mx = fmaxf(mx, (float)cv[t]);
    }
    red[tid] = mx;
    __syncthreads();
    float m2 = red[m * 8];
#pragma unroll
    for (int p = 1; p < 8; ++p) m2 = fmaxf(m2, red[m * 8 + p]);
    __syncthreads();
    float sum = 0.0f;
#pragma unroll
    for (int ch = 0; ch < 8; ++ch) {
      v8h cv = *(const v8h*)&sc[m][i0 + ch * 8];
#pragma unroll
      for (int t = 0; t < 8; ++t) {
        const float e = __expf((float)cv[t] - m2);
        sum += e;
        cv[t] = (_Float16)e;
      }
      *(v8h*)&sc[m][i0 + ch * 8] = cv;
    }
    red[tid] = sum;
    __syncthreads();
    float s2 = 0.0f;
#pragma unroll
    for (int p = 0; p < 8; ++p) s2 += red[m * 8 + p];
    const float inv = 1.0f / s2;
    __syncthreads();
#pragma unroll
    for (int ch = 0; ch < 8; ++ch) {
      v8h cv = *(const v8h*)&sc[m][i0 + ch * 8];
#pragma unroll
      for (int t = 0; t < 8; ++t) cv[t] = (_Float16)((float)cv[t] * inv);
      *(v8h*)&sc[m][i0 + ch * 8] = cv;
    }
    __syncthreads();
  }

  // ---- phase 3: ctx = P(16x512) * V(512x64) --------------------------------
  // kA reused to hold V^T [d][kk] per 64-row chunk; wave w owns d-cols w*16.
  v8f cacc = kZero8;
  for (int kt = 0; kt < Sc; kt += 64) {
#pragma unroll
    for (int i = 0; i < 8; ++i) {                  // V tile: 1024 float4
      const int e = (tid + i * 128) << 2;
      const int kk = e >> 6, d = e & 63;
      const float4 f = *(const float4*)&vp[(size_t)(kt + kk) * Hc + d];
      kA[d + 0][kk] = (_Float16)f.x;               // transpose into V^T
      kA[d + 1][kk] = (_Float16)f.y;
      kA[d + 2][kk] = (_Float16)f.z;
      kA[d + 3][kk] = (_Float16)f.w;
    }
    __syncthreads();
#pragma unroll
    for (int kc = 0; kc < 2; ++kc) {
      const v16h fa = frag_a(&sc[0][0], 520, 0, kt + kc * 32);
      const v16h fb = frag_bT(&kA[0][0], 72, kc * 32, wave * 16);
      cacc = wmma32(fa, fb, cacc);
    }
    __syncthreads();
  }

  float* cp = ctx + (size_t)bb * Sc * Hc + hh * DHc;
  {
    const int dcol = wave * 16 + (lane & 15);
#pragma unroll
    for (int r = 0; r < 8; ++r)
      cp[(size_t)(q0 + rb + r) * Hc + dcol] = cacc[r];
  }
}

// ---------------------------------------------------------------------------
// Embedding gather
// ---------------------------------------------------------------------------
__global__ __launch_bounds__(128) void embed_kernel(
    const int* __restrict__ x, const float* __restrict__ we,
    const float* __restrict__ pe, const float* __restrict__ te,
    float* __restrict__ out) {
  const int row = blockIdx.x;
  const int s   = row & (Sc - 1);
  const int tok = x[row];
  for (int i = threadIdx.x * 4; i < Hc; i += 128 * 4) {
    const float4 a = *(const float4*)&we[(size_t)tok * Hc + i];
    const float4 b = *(const float4*)&pe[(size_t)s * Hc + i];
    const float4 c = *(const float4*)&te[i];
    float4 o;
    o.x = a.x + b.x + c.x; o.y = a.y + b.y + c.y;
    o.z = a.z + b.z + c.z; o.w = a.w + b.w + c.w;
    *(float4*)&out[(size_t)row * Hc + i] = o;
  }
}

// ---------------------------------------------------------------------------
// LayerNorm over H=768, one row per block
// ---------------------------------------------------------------------------
__global__ __launch_bounds__(128) void ln_kernel(
    const float* __restrict__ in, float* __restrict__ out,
    const float* __restrict__ g, const float* __restrict__ b) {
  __shared__ float red[128];
  const int row = blockIdx.x, tid = threadIdx.x;
  const float* p = in + (size_t)row * Hc;

  float s = 0.0f;
  for (int i = tid; i < Hc; i += 128) s += p[i];
  red[tid] = s;
  __syncthreads();
  for (int off = 64; off > 0; off >>= 1) {
    if (tid < off) red[tid] += red[tid + off];
    __syncthreads();
  }
  const float mean = red[0] * (1.0f / Hc);
  __syncthreads();

  float vs = 0.0f;
  for (int i = tid; i < Hc; i += 128) {
    const float d = p[i] - mean;
    vs += d * d;
  }
  red[tid] = vs;
  __syncthreads();
  for (int off = 64; off > 0; off >>= 1) {
    if (tid < off) red[tid] += red[tid + off];
    __syncthreads();
  }
  const float inv = rsqrtf(red[0] * (1.0f / Hc) + 1e-12f);
  float* o = out + (size_t)row * Hc;
  for (int i = tid; i < Hc; i += 128) o[i] = (p[i] - mean) * inv * g[i] + b[i];
}

// ---------------------------------------------------------------------------
// Classifier on last token: out[b,v] = h[b,S-1,:].wc[:,v] + bc[v]
// ---------------------------------------------------------------------------
__global__ __launch_bounds__(128) void cls_kernel(
    const float* __restrict__ h, const float* __restrict__ wc,
    const float* __restrict__ bc, float* __restrict__ out) {
  __shared__ float sA[Bc][Hc];
  for (int i = threadIdx.x; i < Bc * Hc; i += 128) {
    const int bb = i / Hc, kk = i % Hc;
    sA[bb][kk] = h[((size_t)bb * Sc + (Sc - 1)) * Hc + kk];
  }
  __syncthreads();
  const int vcol = blockIdx.x * 128 + threadIdx.x;
  if (vcol >= Vc) return;
  float acc[Bc];
#pragma unroll
  for (int bb = 0; bb < Bc; ++bb) acc[bb] = bc[vcol];
  for (int kk = 0; kk < Hc; ++kk) {
    const float w = wc[(size_t)kk * Vc + vcol];
#pragma unroll
    for (int bb = 0; bb < Bc; ++bb) acc[bb] += sA[bb][kk] * w;
  }
#pragma unroll
  for (int bb = 0; bb < Bc; ++bb) out[(size_t)bb * Vc + vcol] = acc[bb];
}

// ---------------------------------------------------------------------------
// Host orchestration
// ---------------------------------------------------------------------------
extern "C" void kernel_launch(void* const* d_in, const int* in_sizes, int n_in,
                              void* d_out, int out_size, void* d_ws, size_t ws_size,
                              hipStream_t stream) {
  (void)in_sizes; (void)n_in; (void)out_size; (void)ws_size;

  const int*   x      = (const int*)  d_in[0];
  const float* we     = (const float*)d_in[1];
  const float* pe     = (const float*)d_in[2];
  const float* te     = (const float*)d_in[3];
  const float* eg     = (const float*)d_in[4];
  const float* ebi    = (const float*)d_in[5];
  const float* wq0    = (const float*)d_in[6];
  const float* bq0    = (const float*)d_in[7];
  const float* wk0    = (const float*)d_in[8];
  const float* bk0    = (const float*)d_in[9];
  const float* wv0    = (const float*)d_in[10];
  const float* bv0    = (const float*)d_in[11];
  const float* dist0  = (const float*)d_in[12];
  const float* wo0    = (const float*)d_in[13];
  const float* bo0    = (const float*)d_in[14];
  const float* l1g0   = (const float*)d_in[15];
  const float* l1b0   = (const float*)d_in[16];
  const float* wi0    = (const float*)d_in[17];
  const float* bi0    = (const float*)d_in[18];
  const float* wf0    = (const float*)d_in[19];
  const float* bf0    = (const float*)d_in[20];
  const float* l2g0   = (const float*)d_in[21];
  const float* l2b0   = (const float*)d_in[22];
  const float* wc     = (const float*)d_in[23];
  const float* bc     = (const float*)d_in[24];
  float* out = (float*)d_out;

  const size_t sz = (size_t)Mrows * Hc;
  float* ws = (float*)d_ws;
  float* h  = ws;
  float* qb = h  + sz;
  float* kb = qb + sz;
  float* vb = kb + sz;
  float* cb = vb + sz;
  float* t0 = cb + sz;
  float* ff = t0 + sz;                 // Mrows * DF

  embed_kernel<<<Mrows, 128, 0, stream>>>(x, we, pe, te, t0);
  ln_kernel<<<Mrows, 128, 0, stream>>>(t0, h, eg, ebi);

  const dim3 blk(128);
  const dim3 gH(Hc / 64, Mrows / 64);
  const dim3 gF(DFc / 64, Mrows / 64);
  const dim3 gA(Sc / 16, NHc, Bc);

  for (int l = 0; l < Lc; ++l) {
    const size_t oHH = (size_t)l * Hc * Hc;
    const size_t oH  = (size_t)l * Hc;
    const size_t oHF = (size_t)l * Hc * DFc;
    const size_t oF  = (size_t)l * DFc;
    const float* dl  = dist0 + (size_t)l * (2 * Sc - 1) * DHc;

    gemm_wmma_kernel<0><<<gH, blk, 0, stream>>>(h, wq0 + oHH, bq0 + oH, nullptr,
                                                qb, Mrows, Hc, Hc);
    gemm_wmma_kernel<0><<<gH, blk, 0, stream>>>(h, wk0 + oHH, bk0 + oH, nullptr,
                                                kb, Mrows, Hc, Hc);
    gemm_wmma_kernel<0><<<gH, blk, 0, stream>>>(h, wv0 + oHH, bv0 + oH, nullptr,
                                                vb, Mrows, Hc, Hc);

    attn_kernel<<<gA, blk, 0, stream>>>(qb, kb, vb, dl, x, cb);

    gemm_wmma_kernel<2><<<gH, blk, 0, stream>>>(cb, wo0 + oHH, bo0 + oH, h,
                                                t0, Mrows, Hc, Hc);
    ln_kernel<<<Mrows, 128, 0, stream>>>(t0, h, l1g0 + oH, l1b0 + oH);

    gemm_wmma_kernel<1><<<gF, blk, 0, stream>>>(h, wi0 + oHF, bi0 + oF, nullptr,
                                                ff, Mrows, DFc, Hc);
    gemm_wmma_kernel<2><<<gH, blk, 0, stream>>>(ff, wf0 + oHF, bf0 + oH, h,
                                                t0, Mrows, Hc, DFc);
    ln_kernel<<<Mrows, 128, 0, stream>>>(t0, h, l2g0 + oH, l2b0 + oH);
  }

  cls_kernel<<<(Vc + 127) / 128, 128, 0, stream>>>(h, wc, bc, out);
}
